// TriangularAttention_61667140436340
// MI455X (gfx1250) — matrix operand order repaired and checked by reading, not compiled, not run on hardware
//
#include <hip/hip_runtime.h>

// ---------------- problem constants ----------------
#define B_  2
#define S_  48
#define N_  2304        // S*S
#define D_  256
#define H_  8
#define DH_ 32
#define M_  (B_*N_)     // 4608 rows

typedef __attribute__((ext_vector_type(16))) _Float16 v16h;
typedef __attribute__((ext_vector_type(8)))  _Float16 v8h;
typedef __attribute__((ext_vector_type(8)))  float    v8f;

// ---------------- helpers ----------------
__device__ __forceinline__ v16h cat8(v8h lo, v8h hi) {
    return __builtin_shufflevector(lo, hi, 0,1,2,3,4,5,6,7,8,9,10,11,12,13,14,15);
}

// A-operand (16x32 f16): lane<16 -> row=lane,   K 0-15
//                        lane>=16 -> row=lane-16, K 16-31
__device__ __forceinline__ v16h loadA(const _Float16* rowk0, int lane) {
    const int off = (lane < 16) ? 0 : 16;
    v8h lo = *(const v8h*)(rowk0 + off);
    v8h hi = *(const v8h*)(rowk0 + off + 8);
    return cat8(lo, hi);
}

__device__ __forceinline__ v8f wmma_f16(v16h a, v16h b, v8f c) {
    return __builtin_amdgcn_wmma_f32_16x16x32_f16(
        /*neg_a=*/false, a, /*neg_b=*/false, b,
        /*c_mod=*/(short)0, c, /*reuse_a=*/false, /*reuse_b=*/false);
}

// ---------------- conversion kernels ----------------
__global__ void k_cvt_f16(const float* __restrict__ in, _Float16* __restrict__ out, int n) {
    int i = blockIdx.x * blockDim.x + threadIdx.x;
    if (i < n) out[i] = (_Float16)in[i];
}

// WT[c][k] = (f16) W[k][c]   (256x256)
__global__ void k_transpose_f16(const float* __restrict__ W, _Float16* __restrict__ WT) {
    int k = blockIdx.x;
    int c = threadIdx.x;
    WT[c * D_ + k] = (_Float16)W[k * D_ + c];
}

// ---------------- projection GEMM: 16x64 tile per wave ----------------
// A: (M_ x 256) f16 row-major. WT: (256 outcols x 256) f16 (pre-transposed).
// mode 0: out f16 in (B,H,N,DH) layout   (q, k)
// mode 1: out f16 in (B,H,DH,N) layout   (v transposed for PV B-operand)
// mode 2: out f32 row-major (final projection, + bias)
__global__ void __launch_bounds__(256) k_gemm(const _Float16* __restrict__ A,
                                              const _Float16* __restrict__ WT,
                                              const float* __restrict__ bias,
                                              void* __restrict__ dst, int mode) {
    const int lane = threadIdx.x & 31;
    const int w    = blockIdx.x * (blockDim.x >> 5) + (threadIdx.x >> 5);
    const int tm   = w >> 2;            // 288 row tiles (M/16)
    const int tn   = w & 3;             // 4 col tiles (256/64)
    const int row  = tm * 16 + (lane & 15);
    const int c0   = tn * 64 + (lane & 15);

    const _Float16* ar = A + (size_t)row * D_;
    const _Float16* bc0 = WT + (size_t)(c0 +  0) * D_;
    const _Float16* bc1 = WT + (size_t)(c0 + 16) * D_;
    const _Float16* bc2 = WT + (size_t)(c0 + 32) * D_;
    const _Float16* bc3 = WT + (size_t)(c0 + 48) * D_;
    const int boff = (lane < 16) ? 0 : 16;

    v8f acc0{}, acc1{}, acc2{}, acc3{};
    #pragma unroll
    for (int k0 = 0; k0 < D_; k0 += 32) {
        v16h a  = loadA(ar + k0, lane);             // reused across 4 WMMAs
        v16h b0 = *(const v16h*)(bc0 + k0 + boff);
        v16h b1 = *(const v16h*)(bc1 + k0 + boff);
        v16h b2 = *(const v16h*)(bc2 + k0 + boff);
        v16h b3 = *(const v16h*)(bc3 + k0 + boff);
        acc0 = wmma_f16(a, b0, acc0);
        acc1 = wmma_f16(a, b1, acc1);
        acc2 = wmma_f16(a, b2, acc2);
        acc3 = wmma_f16(a, b3, acc3);
    }

    const int rbase = tm * 16 + ((lane < 16) ? 0 : 8);   // C/D layout rows
    v8f accs[4] = {acc0, acc1, acc2, acc3};

    #pragma unroll
    for (int j = 0; j < 4; ++j) {
        const int col = c0 + j * 16;
        const float bv = bias[col];
        if (mode == 2) {
            float* out = (float*)dst;
            #pragma unroll
            for (int r = 0; r < 8; ++r)
                out[(size_t)(rbase + r) * D_ + col] = accs[j][r] + bv;
        } else {
            _Float16* out = (_Float16*)dst;
            const int h = col / DH_, dh = col % DH_;
            #pragma unroll
            for (int r = 0; r < 8; ++r) {
                int grow = rbase + r;
                int b = grow / N_, n = grow % N_;
                float val = accs[j][r] + bv;
                if (mode == 0)
                    out[(((size_t)(b * H_ + h)) * N_ + n) * DH_ + dh] = (_Float16)val;
                else
                    out[(((size_t)(b * H_ + h)) * DH_ + dh) * N_ + n] = (_Float16)val;
            }
        }
    }
}

// ---------------- flash attention: 16 queries/wave, 64-key blocks ----------------
__global__ void __launch_bounds__(256) k_attn(const _Float16* __restrict__ qb,
                                              const _Float16* __restrict__ kb,
                                              const _Float16* __restrict__ vT,
                                              _Float16* __restrict__ ctx) {
    __shared__ __align__(16) _Float16 ldsP[8][16 * 64];   // 2KB per wave, private

    const int lane = threadIdx.x & 31;
    const int wv   = threadIdx.x >> 5;
    const int w    = blockIdx.x * 8 + wv;
    const int ntiles = N_ / 16;                 // 144
    const int b   = w / (H_ * ntiles);
    const int rem = w % (H_ * ntiles);
    const int h   = rem / ntiles;
    const int nt  = rem % ntiles;

    const _Float16* qbh = qb + ((size_t)(b * H_ + h)) * N_ * DH_;
    const _Float16* kbh = kb + ((size_t)(b * H_ + h)) * N_ * DH_;
    const _Float16* vbh = vT + ((size_t)(b * H_ + h)) * DH_ * N_;

    const int lhalf = (lane < 16) ? 0 : 8;
    const int lc    = lane & 15;

    // Q A-operand, held in registers for the whole key loop
    const int qrow = nt * 16 + lc;
    v16h qa = loadA(qbh + (size_t)qrow * DH_, lane);

    const float scale = 0.17677669529663687f;   // 1/sqrt(32)

    float mrow[8], lrow[8];
    #pragma unroll
    for (int r = 0; r < 8; ++r) { mrow[r] = -1e30f; lrow[r] = 0.f; }
    v8f o0{}, o1{};

    _Float16* lp = &ldsP[wv][0];
    const int koffB = (lane < 16) ? 0 : 16;     // B-operand contraction sub-offset

    for (int m0 = 0; m0 < N_; m0 += 64) {
        // ---- QK^T: 4 key tiles of 16 (dh contraction = 32, one WMMA each) ----
        const _Float16* kp = kbh + (size_t)(m0 + lc) * DH_ + ((lane < 16) ? 0 : 16);
        v16h kB0 = *(const v16h*)(kp);
        v16h kB1 = *(const v16h*)(kp + 16 * DH_);
        v16h kB2 = *(const v16h*)(kp + 32 * DH_);
        v16h kB3 = *(const v16h*)(kp + 48 * DH_);
        v8f s0{}, s1{}, s2{}, s3{};
        s0 = wmma_f16(qa, kB0, s0);
        s1 = wmma_f16(qa, kB1, s1);
        s2 = wmma_f16(qa, kB2, s2);
        s3 = wmma_f16(qa, kB3, s3);

        // ---- online softmax per row (row r lives in VGPR r across a 16-lane half)
        #pragma unroll
        for (int r = 0; r < 8; ++r) {
            float a0 = s0[r] * scale;
            float a1 = s1[r] * scale;
            float a2 = s2[r] * scale;
            float a3 = s3[r] * scale;
            float bm = fmaxf(fmaxf(a0, a1), fmaxf(a2, a3));
            #pragma unroll
            for (int msk = 1; msk < 16; msk <<= 1) bm = fmaxf(bm, __shfl_xor(bm, msk, 32));
            float mn    = fmaxf(mrow[r], bm);
            float alpha = __expf(mrow[r] - mn);
            mrow[r] = mn;
            float p0 = __expf(a0 - mn);
            float p1 = __expf(a1 - mn);
            float p2 = __expf(a2 - mn);
            float p3 = __expf(a3 - mn);
            float rs = (p0 + p1) + (p2 + p3);
            #pragma unroll
            for (int msk = 1; msk < 16; msk <<= 1) rs += __shfl_xor(rs, msk, 32);
            lrow[r] = lrow[r] * alpha + rs;
            o0[r] *= alpha;
            o1[r] *= alpha;
            // stage P (C/D layout -> row-major f16 in LDS, row stride 64)
            _Float16* lrp = lp + (lhalf + r) * 64 + lc;
            lrp[0]  = (_Float16)p0;
            lrp[16] = (_Float16)p1;
            lrp[32] = (_Float16)p2;
            lrp[48] = (_Float16)p3;
        }

        // ---- P @ V : two 32-key chunks x two d tiles ----
        {
            const int prow = lc;
            const int koffA = (lane < 16) ? 0 : 16;
            // P A-operand, chunk 0 (keys m0..m0+31) and chunk 1 (m0+32..m0+63)
            v8h plo0 = *(const v8h*)(lp + prow * 64 + koffA);
            v8h phi0 = *(const v8h*)(lp + prow * 64 + koffA + 8);
            v8h plo1 = *(const v8h*)(lp + prow * 64 + 32 + koffA);
            v8h phi1 = *(const v8h*)(lp + prow * 64 + 32 + koffA + 8);
            v16h pa0 = cat8(plo0, phi0);
            v16h pa1 = cat8(plo1, phi1);
            // V B-operands (transposed layout: key index contiguous per lane)
            const _Float16* vp = vbh + (size_t)lc * N_ + m0 + koffB;
            v16h vB00 = *(const v16h*)(vp);                       // chunk0, d 0-15
            v16h vB01 = *(const v16h*)(vp + (size_t)16 * N_);     // chunk0, d 16-31
            v16h vB10 = *(const v16h*)(vp + 32);                  // chunk1, d 0-15
            v16h vB11 = *(const v16h*)(vp + (size_t)16 * N_ + 32);// chunk1, d 16-31
            o0 = wmma_f16(pa0, vB00, o0);
            o1 = wmma_f16(pa0, vB01, o1);
            o0 = wmma_f16(pa1, vB10, o0);
            o1 = wmma_f16(pa1, vB11, o1);
        }
    }

    // normalize and write ctx (f16, (B,N,D) with col = h*32 + d)
    #pragma unroll
    for (int r = 0; r < 8; ++r) {
        float inv = 1.0f / lrow[r];
        int n = nt * 16 + lhalf + r;
        size_t rowoff = ((size_t)b * N_ + n) * D_ + h * DH_;
        ctx[rowoff + lc]      = (_Float16)(o0[r] * inv);
        ctx[rowoff + lc + 16] = (_Float16)(o1[r] * inv);
    }
}

// ---------------- launcher ----------------
extern "C" void kernel_launch(void* const* d_in, const int* in_sizes, int n_in,
                              void* d_out, int out_size, void* d_ws, size_t ws_size,
                              hipStream_t stream) {
    (void)in_sizes; (void)n_in; (void)out_size; (void)ws_size;
    const float* x  = (const float*)d_in[0];
    const float* Wq = (const float*)d_in[1];
    const float* bq = (const float*)d_in[2];
    const float* Wk = (const float*)d_in[3];
    const float* bk = (const float*)d_in[4];
    const float* Wv = (const float*)d_in[5];
    const float* bv = (const float*)d_in[6];
    const float* Wo = (const float*)d_in[7];
    const float* bo = (const float*)d_in[8];
    float* out = (float*)d_out;

    char* ws = (char*)d_ws;
    _Float16* xb   = (_Float16*)(ws);                 // 4608*256 f16 = 2359296 B
    _Float16* WqT  = (_Float16*)(ws + 2359296);       // 4 x 65536 f16
    _Float16* WkT  = WqT + 65536;
    _Float16* WvT  = WkT + 65536;
    _Float16* WoT  = WvT + 65536;
    _Float16* qb   = (_Float16*)(ws + 2883584);       // (B,H,N,DH) f16
    _Float16* kbuf = (_Float16*)(ws + 5242880);       // (B,H,N,DH) f16
    _Float16* vTb  = (_Float16*)(ws + 7602176);       // (B,H,DH,N) f16
    _Float16* ctxb = (_Float16*)(ws + 9961472);       // (B,N,D)    f16
    // total workspace: 12,320,768 bytes

    const int nx = M_ * D_;                           // 1,179,648
    k_cvt_f16<<<(nx + 255) / 256, 256, 0, stream>>>(x, xb, nx);
    k_transpose_f16<<<D_, D_, 0, stream>>>(Wq, WqT);
    k_transpose_f16<<<D_, D_, 0, stream>>>(Wk, WkT);
    k_transpose_f16<<<D_, D_, 0, stream>>>(Wv, WvT);
    k_transpose_f16<<<D_, D_, 0, stream>>>(Wo, WoT);

    const int gemm_blocks = (M_ / 16) * (D_ / 64) / 8;   // 144 blocks, 8 waves each
    k_gemm<<<gemm_blocks, 256, 0, stream>>>(xb, WqT, bq, (void*)qb,   0);
    k_gemm<<<gemm_blocks, 256, 0, stream>>>(xb, WkT, bk, (void*)kbuf, 0);
    k_gemm<<<gemm_blocks, 256, 0, stream>>>(xb, WvT, bv, (void*)vTb,  1);

    const int attn_blocks = (B_ * H_ * (N_ / 16)) / 8;   // 288
    k_attn<<<attn_blocks, 256, 0, stream>>>(qb, kbuf, vTb, ctxb);

    k_gemm<<<gemm_blocks, 256, 0, stream>>>(ctxb, WoT, bo, (void*)out, 2);
}
